// QKVDotProductAttention_29738353557988
// MI455X (gfx1250) — matrix operand order repaired and checked by reading, not compiled, not run on hardware
//
#include <hip/hip_runtime.h>

// ---------------------------------------------------------------------------
// QKV dot-product attention for MI455X (gfx1250), wave32 + WMMA f16->f32.
// B=8, TQ=1024, TK=2048, F=1024.
//
// Pipeline:
//   1) Qh  = f16( inputs   @ Wq^T + bq )            [B*TQ, F]
//   2) Kh  = f16( features @ Wk^T + bk )            [B*TK, F]
//   3) Vth = f16( features @ Wv^T + bv ) transposed [B][F][TK]
//   4) raw scores = scale * Qh @ Kh^T, masked -> weights region of d_out (f32)
//      (double-buffered LDS, global_load_async_to_lds_b128 staging)
//   5) row softmax in place (f32) + f16 copy of weights into workspace
//   6) context = W16 @ Vt -> context region of d_out (f32)  (same async scheme)
// ---------------------------------------------------------------------------

typedef __attribute__((ext_vector_type(16))) _Float16 v16h;
typedef __attribute__((ext_vector_type(8)))  _Float16 h8;
typedef __attribute__((ext_vector_type(4)))  _Float16 h4;
typedef __attribute__((ext_vector_type(8)))  float    v8f;
typedef __attribute__((ext_vector_type(4)))  float    f4;

#define BLK_M 128
#define BLK_N 128
#define BLK_K 32
#define LDS_STRIDE 40   // 32 halves + 8 halves pad: breaks bank conflicts on b128 frag loads
#define SLAB (BLK_M * LDS_STRIDE)
#define NTHREADS 256    // 8 wave32s: wave grid 2 (M) x 4 (N), each wave -> 64x32 output

// ---- LDS staging ----------------------------------------------------------

// f32 -> f16 conversion staging (projection GEMMs): load/convert/store.
__device__ inline void stage_f32(const float* __restrict__ g, int ld, int row0, int k0,
                                 _Float16* __restrict__ s) {
  const int t = threadIdx.x;
  const int row  = t >> 1;          // 0..127
  const int half = (t & 1) * 16;    // 0 or 16
  const float* src = g + (size_t)(row0 + row) * ld + k0 + half;
  _Float16* dst = s + row * LDS_STRIDE + half;
#pragma unroll
  for (int j = 0; j < 4; ++j) {
    f4 v = *(const f4*)(src + j * 4);                       // global_load_b128
    h4 h = { (_Float16)v.x, (_Float16)v.y, (_Float16)v.z, (_Float16)v.w };
    *(h4*)(dst + j * 4) = h;                                // ds_store_b64
  }
}

// Async DMA staging (f16 slabs): per-lane global_load_async_to_lds_b128.
// each lane copies 16 bytes from global memory to LDS, tracked by ASYNCcnt.
// Low 32 bits of a flat pointer into LDS are the LDS byte offset (aperture rule).
__device__ inline void stage_async_f16(const _Float16* __restrict__ g, int ld,
                                       int row0, int k0, _Float16* s) {
  const int t = threadIdx.x;
  const int row  = t >> 1;          // 0..127
  const int half = (t & 1) * 16;    // 0 or 16
  const unsigned long long src =
      (unsigned long long)(g + (size_t)(row0 + row) * ld + k0 + half);
  const unsigned int dst = (unsigned int)(size_t)(s + row * LDS_STRIDE + half);
  asm volatile("global_load_async_to_lds_b128 %0, %1, off"
               :: "v"(dst), "v"(src) : "memory");
  asm volatile("global_load_async_to_lds_b128 %0, %1, off offset:16"
               :: "v"(dst), "v"(src) : "memory");
}

__device__ inline void wait_async0() {
  asm volatile("s_wait_asynccnt 0" ::: "memory");
}

template <typename T>
__device__ inline void prefetch_slab(const T* __restrict__ g, int ld, int row0, int k0) {
  const int t = threadIdx.x;
  const int row  = t >> 1;
  const int half = (t & 1) * 16;
  __builtin_prefetch((const void*)(g + (size_t)(row0 + row) * ld + k0 + half), 0, 1);
}

// ---- WMMA fragment loads (layouts per CDNA5 ISA 7.12.2) -------------------
// A (16x32 f16): lane<16 holds row M=lane&15, K = {cb..cb+7, 16+cb..23+cb}, cb = 8*(lane>=16)
__device__ inline v16h load_a_frag(const _Float16* __restrict__ s, int row_base, int lane) {
  const int r  = row_base + (lane & 15);
  const int cb = (lane >> 4) * 8;
  const _Float16* p = s + r * LDS_STRIDE + cb;
  h8 lo = *(const h8*)(p);        // ds_load_b128
  h8 hi = *(const h8*)(p + 16);   // ds_load_b128
  return __builtin_shufflevector(lo, hi, 0,1,2,3,4,5,6,7,8,9,10,11,12,13,14,15);
}
// B (32x16 f16, fed as (N,K)-row-major): lane&15 = N, K = {cb..cb+7, 16+cb..23+cb}, cb = 8*(lane>=16)
__device__ inline v16h load_b_frag(const _Float16* __restrict__ s, int col_base, int lane) {
  const int n  = col_base + (lane & 15);
  const int cb = (lane >> 4) * 8;
  const _Float16* p = s + n * LDS_STRIDE + cb;
  h8 lo = *(const h8*)(p);
  h8 hi = *(const h8*)(p + 16);
  return __builtin_shufflevector(lo, hi, 0,1,2,3,4,5,6,7,8,9,10,11,12,13,14,15);
}

// One K-step of the 128x128 macro tile for this wave: 4x2 WMMAs.
__device__ inline void mma_tile(const _Float16* __restrict__ sA, const _Float16* __restrict__ sB,
                                int wm, int wn, int lane, v8f acc[4][2]) {
  v16h bfrag[2];
#pragma unroll
  for (int j = 0; j < 2; ++j) bfrag[j] = load_b_frag(sB, wn * 32 + j * 16, lane);
#pragma unroll
  for (int i = 0; i < 4; ++i) {
    v16h afrag = load_a_frag(sA, wm * 64 + i * 16, lane);
#pragma unroll
    for (int j = 0; j < 2; ++j) {
      acc[i][j] = __builtin_amdgcn_wmma_f32_16x16x32_f16(
          false, afrag, false, bfrag[j], (short)0, acc[i][j], false, false);
    }
  }
}

// ---------------------------------------------------------------------------
// Kernel 1-3: projection GEMM.  out[m,n] = sum_k X[m,k]*W[n,k] + bias[n] -> f16
// TRANSPOSE=true writes per-batch transposed: out[b][n][m%tpb]  (for V^T).
// ---------------------------------------------------------------------------
template <bool TRANSPOSE>
__global__ void __launch_bounds__(NTHREADS)
proj_gemm(const float* __restrict__ X, const float* __restrict__ W,
          const float* __restrict__ bias, _Float16* __restrict__ out,
          int M, int N, int K, int tpb /* tokens per batch (for TRANSPOSE) */) {
  __shared__ _Float16 sA[SLAB];
  __shared__ _Float16 sB[SLAB];

  const int m0 = blockIdx.y * BLK_M;
  const int n0 = blockIdx.x * BLK_N;
  const int lane = threadIdx.x & 31;
  const int wid  = threadIdx.x >> 5;
  const int wm = wid >> 2;   // 0..1
  const int wn = wid & 3;    // 0..3

  v8f acc[4][2] = {};

  for (int k0 = 0; k0 < K; k0 += BLK_K) {
    stage_f32(X, K, m0, k0, sA);
    stage_f32(W, K, n0, k0, sB);
    __syncthreads();
    if (k0 + BLK_K < K) {                 // gfx1250 global_prefetch_b8
      prefetch_slab(X, K, m0, k0 + BLK_K);
      prefetch_slab(W, K, n0, k0 + BLK_K);
    }
    mma_tile(sA, sB, wm, wn, lane, acc);
    __syncthreads();
  }

  const int bb = TRANSPOSE ? (m0 / tpb) : 0;       // whole block is within one batch
  const int mloc0 = TRANSPOSE ? (m0 - bb * tpb) : 0;

#pragma unroll
  for (int i = 0; i < 4; ++i) {
#pragma unroll
    for (int j = 0; j < 2; ++j) {
      const int col = n0 + wn * 32 + j * 16 + (lane & 15);
      const float bval = bias[col];
      const int rbase = wm * 64 + i * 16 + ((lane >> 4) << 3);
#pragma unroll
      for (int g = 0; g < 8; ++g) {
        const float val = acc[i][j][g] + bval;
        if (!TRANSPOSE) {
          const int r = m0 + rbase + g;
          out[(size_t)r * N + col] = (_Float16)val;
        } else {
          const int kl = mloc0 + rbase + g;
          out[((size_t)bb * N + col) * tpb + kl] = (_Float16)val;
        }
      }
    }
  }
}

// ---------------------------------------------------------------------------
// Kernel 4: raw masked scores.  S[b,q,k] = mask ? scale * Q_b[q,:].K_b[k,:] : -inf
// Double-buffered LDS with async DMA staging: one barrier per k-step; the
// async copy of slab k+1 overlaps the 8 WMMAs on slab k.
// ---------------------------------------------------------------------------
__global__ void __launch_bounds__(NTHREADS)
attn_scores(const _Float16* __restrict__ Q, const _Float16* __restrict__ Kp,
            const unsigned char* __restrict__ mask, const float* __restrict__ scale_p,
            float* __restrict__ S, int TQn, int TKn, int Fn) {
  __shared__ _Float16 sA[2][SLAB];
  __shared__ _Float16 sB[2][SLAB];

  const int b  = blockIdx.z;
  const int m0 = blockIdx.y * BLK_M;   // query rows
  const int n0 = blockIdx.x * BLK_N;   // key cols
  const int lane = threadIdx.x & 31;
  const int wid  = threadIdx.x >> 5;
  const int wm = wid >> 2, wn = wid & 3;

  const _Float16* Qb = Q  + (size_t)b * TQn * Fn;
  const _Float16* Kb = Kp + (size_t)b * TKn * Fn;
  const unsigned char* mb = mask + (size_t)b * TQn * TKn;   // row 0: mask is q-broadcast
  float* Sb = S + (size_t)b * TQn * TKn;

  v8f acc[4][2] = {};
  int cur = 0;
  stage_async_f16(Qb, Fn, m0, 0, sA[0]);
  stage_async_f16(Kb, Fn, n0, 0, sB[0]);
  for (int k0 = 0; k0 < Fn; k0 += BLK_K) {
    wait_async0();          // own-wave async copies into sX[cur] complete
    __syncthreads();        // all waves' copies visible; all done reading sX[cur^1]
    if (k0 + BLK_K < Fn) {
      stage_async_f16(Qb, Fn, m0, k0 + BLK_K, sA[cur ^ 1]);
      stage_async_f16(Kb, Fn, n0, k0 + BLK_K, sB[cur ^ 1]);
    }
    mma_tile(sA[cur], sB[cur], wm, wn, lane, acc);
    cur ^= 1;
  }

  const float scl = *scale_p;
#pragma unroll
  for (int i = 0; i < 4; ++i) {
#pragma unroll
    for (int j = 0; j < 2; ++j) {
      const int col = n0 + wn * 32 + j * 16 + (lane & 15);
      const bool valid = mb[col] != 0;
      const int rbase = m0 + wm * 64 + i * 16 + ((lane >> 4) << 3);
#pragma unroll
      for (int g = 0; g < 8; ++g) {
        const float v = valid ? acc[i][j][g] * scl : -__builtin_inff();
        Sb[(size_t)(rbase + g) * TKn + col] = v;
      }
    }
  }
}

// ---------------------------------------------------------------------------
// Kernel 5: row softmax, one wave per 2048-wide row, register resident.
// Overwrites S in place with f32 weights, writes f16 copy to W16.
// ---------------------------------------------------------------------------
__global__ void __launch_bounds__(NTHREADS)
softmax_rows(float* __restrict__ S, _Float16* __restrict__ W16, int nrows, int TKn) {
  const int lane = threadIdx.x & 31;
  const int wid  = threadIdx.x >> 5;
  const int row  = blockIdx.x * 8 + wid;
  if (row >= nrows) return;

  float* rowp = S + (size_t)row * TKn;
  f4 v[16];                                     // 64 floats / lane * 32 lanes = 2048
  float m = -__builtin_inff();
#pragma unroll
  for (int i = 0; i < 16; ++i) {
    v[i] = *(const f4*)(rowp + (i * 32 + lane) * 4);
    m = fmaxf(m, fmaxf(fmaxf(v[i].x, v[i].y), fmaxf(v[i].z, v[i].w)));
  }
#pragma unroll
  for (int off = 16; off >= 1; off >>= 1) m = fmaxf(m, __shfl_xor(m, off, 32));

  float sum = 0.f;
#pragma unroll
  for (int i = 0; i < 16; ++i) {
#pragma unroll
    for (int c = 0; c < 4; ++c) {
      const float e = __expf(v[i][c] - m);      // exp(-inf - m) == 0 for masked keys
      v[i][c] = e;
      sum += e;
    }
  }
#pragma unroll
  for (int off = 16; off >= 1; off >>= 1) sum += __shfl_xor(sum, off, 32);
  const float inv = 1.0f / sum;

  _Float16* w16p = W16 + (size_t)row * TKn;
#pragma unroll
  for (int i = 0; i < 16; ++i) {
    const int idx = (i * 32 + lane) * 4;
    f4 w = v[i] * inv;
    *(f4*)(rowp + idx) = w;                                   // f32 weights output
    h4 h = { (_Float16)w.x, (_Float16)w.y, (_Float16)w.z, (_Float16)w.w };
    *(h4*)(w16p + idx) = h;                                   // f16 copy for PV GEMM
  }
}

// ---------------------------------------------------------------------------
// Kernel 6: context = W16 @ V  (V stored transposed: Vt[b][d][k], (N,K)-row-major)
// Same double-buffered async staging as the scores kernel.
// ---------------------------------------------------------------------------
__global__ void __launch_bounds__(NTHREADS)
attn_context(const _Float16* __restrict__ Wp, const _Float16* __restrict__ Vt,
             float* __restrict__ C, int TQn, int TKn, int Fn) {
  __shared__ _Float16 sA[2][SLAB];
  __shared__ _Float16 sB[2][SLAB];

  const int b  = blockIdx.z;
  const int m0 = blockIdx.y * BLK_M;   // query rows
  const int n0 = blockIdx.x * BLK_N;   // feature cols
  const int lane = threadIdx.x & 31;
  const int wid  = threadIdx.x >> 5;
  const int wm = wid >> 2, wn = wid & 3;

  const _Float16* Wb = Wp + (size_t)b * TQn * TKn;
  const _Float16* Vb = Vt + (size_t)b * Fn * TKn;
  float* Cb = C + (size_t)b * TQn * Fn;

  v8f acc[4][2] = {};
  int cur = 0;
  stage_async_f16(Wb, TKn, m0, 0, sA[0]);
  stage_async_f16(Vb, TKn, n0, 0, sB[0]);
  for (int k0 = 0; k0 < TKn; k0 += BLK_K) {
    wait_async0();
    __syncthreads();
    if (k0 + BLK_K < TKn) {
      stage_async_f16(Wb, TKn, m0, k0 + BLK_K, sA[cur ^ 1]);
      stage_async_f16(Vb, TKn, n0, k0 + BLK_K, sB[cur ^ 1]);
    }
    mma_tile(sA[cur], sB[cur], wm, wn, lane, acc);
    cur ^= 1;
  }

#pragma unroll
  for (int i = 0; i < 4; ++i) {
#pragma unroll
    for (int j = 0; j < 2; ++j) {
      const int col = n0 + wn * 32 + j * 16 + (lane & 15);
      const int rbase = m0 + wm * 64 + i * 16 + ((lane >> 4) << 3);
#pragma unroll
      for (int g = 0; g < 8; ++g)
        Cb[(size_t)(rbase + g) * Fn + col] = acc[i][j][g];
    }
  }
}

// ---------------------------------------------------------------------------
extern "C" void kernel_launch(void* const* d_in, const int* in_sizes, int n_in,
                              void* d_out, int out_size, void* d_ws, size_t ws_size,
                              hipStream_t stream) {
  const int Bc = 8, TQc = 1024, TKc = 2048, Fc = 1024;

  const float*         inputs   = (const float*)d_in[0];
  const float*         features = (const float*)d_in[1];
  const unsigned char* mask     = (const unsigned char*)d_in[2];  // jax bool -> 1 byte
  const float* Wq = (const float*)d_in[3];
  const float* bq = (const float*)d_in[4];
  const float* Wk = (const float*)d_in[5];
  const float* bk = (const float*)d_in[6];
  const float* Wv = (const float*)d_in[7];
  const float* bv = (const float*)d_in[8];
  const float* scale = (const float*)d_in[9];

  float* ctx = (float*)d_out;                        // [B,TQ,F]
  float* wts = ctx + (size_t)Bc * TQc * Fc;          // [B,1,TQ,TK]

  // workspace: Qh 16MB | Kh 32MB | Vth 32MB | W16 32MB  = 112MB
  _Float16* Qh  = (_Float16*)d_ws;
  _Float16* Kh  = Qh  + (size_t)Bc * TQc * Fc;
  _Float16* Vth = Kh  + (size_t)Bc * TKc * Fc;
  _Float16* W16 = Vth + (size_t)Bc * TKc * Fc;

  const dim3 blk(NTHREADS);

  // 1-3) projections (flattened over batch; inputs/features are contiguous [B*T, F])
  proj_gemm<false><<<dim3(Fc / BLK_N, (Bc * TQc) / BLK_M), blk, 0, stream>>>(
      inputs, Wq, bq, Qh, Bc * TQc, Fc, Fc, TQc);
  proj_gemm<false><<<dim3(Fc / BLK_N, (Bc * TKc) / BLK_M), blk, 0, stream>>>(
      features, Wk, bk, Kh, Bc * TKc, Fc, Fc, TKc);
  proj_gemm<true><<<dim3(Fc / BLK_N, (Bc * TKc) / BLK_M), blk, 0, stream>>>(
      features, Wv, bv, Vth, Bc * TKc, Fc, Fc, TKc);

  // 4) masked raw scores into the weights output region (reused as scratch)
  attn_scores<<<dim3(TKc / BLK_N, TQc / BLK_M, Bc), blk, 0, stream>>>(
      Qh, Kh, mask, scale, wts, TQc, TKc, Fc);

  // 5) row softmax in place + f16 weights copy
  softmax_rows<<<dim3((Bc * TQc) / 8), blk, 0, stream>>>(wts, W16, Bc * TQc, TKc);

  // 6) context
  attn_context<<<dim3(Fc / BLK_N, TQc / BLK_M, Bc), blk, 0, stream>>>(
      W16, Vth, ctx, TQc, TKc, Fc);
}
